// PointSIS_35665408426487
// MI455X (gfx1250) — compile-verified, matrix operand order checked by reading, not run on hardware
//
#include <hip/hip_runtime.h>
#include <hip/hip_bf16.h>

// ---------------------------------------------------------------------------
// PointSIS forward for MI455X (gfx1250).
// bf16 WMMA GEMMs, double-buffered async global->LDS A staging, bf16 producers.
// ---------------------------------------------------------------------------

#define G_GROUPS 32768
#define NB       8
#define GN_ROWS  (G_GROUPS * NB)      // 262144
#define D_MODEL  256
#define D_INNER  512
#define D_STATE  16
#define DT_RANK  16
#define D_CONV   4
#define PATCH_L  1024
#define N_PATCH  (G_GROUPS / PATCH_L) // 32

typedef __attribute__((ext_vector_type(16))) __bf16 v16bf;
typedef __attribute__((ext_vector_type(8)))  float  v8f;

// ---------------- helpers ----------------
__device__ __forceinline__ unsigned short f2bf_bits(float f) {
  unsigned u = __float_as_uint(f);
  unsigned r = u + 0x7FFFu + ((u >> 16) & 1u);   // round-to-nearest-even
  return (unsigned short)(r >> 16);
}
__device__ __forceinline__ __bf16 f2bf(float f) {
  unsigned short s = f2bf_bits(f);
  __bf16 o; __builtin_memcpy(&o, &s, 2); return o;
}
__device__ __forceinline__ float bf2f(__bf16 b) {
  unsigned short s; __builtin_memcpy(&s, &b, 2);
  return __uint_as_float(((unsigned)s) << 16);
}
__device__ __forceinline__ float gelu_f(float x) {
  float x3 = x * x * x;
  return 0.5f * x * (1.0f + tanhf(0.7978845608f * (x + 0.044715f * x3)));
}
__device__ __forceinline__ float silu_f(float x) {
  return x * (1.0f / (1.0f + __expf(-x)));
}

// ---------------- pack f32 -> bf16 (weights) ----------------
__global__ void pack_bf16_k(const float* __restrict__ s, __bf16* __restrict__ d, long n) {
  long i = (long)blockIdx.x * blockDim.x + threadIdx.x;
  if (i < n) d[i] = f2bf(s[i]);
}

// ---------------- WMMA GEMM:  C[M,N] = X[M,K] @ W[N,K]^T (+bias) ------------
// X, W bf16 (K-contiguous). Block = 256 threads = 8 waves (2 M x 4 N). Each
// wave owns a 32x16 output tile (two v8f accumulators, two v_wmma per K-step,
// shared B fragment). A tiles (64x32 bf16) are staged with one
// global_load_async_to_lds_b128 per lane into double-buffered LDS; the next
// tile's DMA overlaps the current tile's WMMAs (s_wait_asynccnt 1).
// Preconditions: M % 64 == 0; K % 16 == 0; rows of X readable up to
// ceil(K/32)*32 (lda >= 32 when K == 16).
#define BM 64
#define BN 64
#define BK 32
#define APAD 48   // LDS row stride in elements (96B; 16B-aligned fragments)

__global__ __launch_bounds__(256)
void wmma_gemm_bf16_k(const __bf16* __restrict__ X, int lda,
                      const __bf16* __restrict__ W,
                      const float* __restrict__ bias,
                      float* __restrict__ C, int ldc,
                      __bf16* __restrict__ Cb, int ldcb,
                      long M, int N, int K) {
  __shared__ __align__(16) __bf16 As[2][BM][APAD];

  const int tid  = threadIdx.x;
  const int wid  = tid >> 5;
  const int lane = tid & 31;
  const int wm   = wid >> 2;           // 0..1  (32-row group)
  const int wn   = wid & 3;            // 0..3  (16-col group)
  const long blockM = (long)blockIdx.y * BM;
  const int  blockN = blockIdx.x * BN;

  const int row16 = lane & 15;
  const int khalf = lane >> 4;         // 0/1

  // staging map: 4 threads per row, 16B (8 bf16) each
  const int lrow = tid >> 2;           // 0..63
  const int lcol = (tid & 3) * 8;      // 0,8,16,24

  v8f acc0 = {0.f, 0.f, 0.f, 0.f, 0.f, 0.f, 0.f, 0.f};
  v8f acc1 = acc0;

  const int col = blockN + wn * 16 + row16;      // output column / W row
  const __bf16* Wrow = W + (size_t)col * K;
  const __bf16* Xrow = X + (size_t)(blockM + lrow) * lda + lcol;

  const __bf16* ldsbase = &As[0][0][0];
  const unsigned off0 =
      (unsigned)((const char*)&As[0][lrow][lcol] - (const char*)ldsbase);
  const unsigned off1 =
      (unsigned)((const char*)&As[1][lrow][lcol] - (const char*)ldsbase);

  // Async DMA issue; ldsbase operand makes As escape so the compiler keeps
  // the subsequent ds_load fragment reads (asm may write LDS).
#define ISSUE_ASYNC(OFF, GP)                                              \
  asm volatile("global_load_async_to_lds_b128 %0, %1, off"                \
               :: "v"(OFF), "v"(GP), "v"(ldsbase) : "memory")

  ISSUE_ASYNC(off0, Xrow);                      // prefill buffer 0
  int buf = 0;

  const bool interior = (blockN + BN <= N) && ((K & (BK - 1)) == 0);

  if (interior) {
    for (int k0 = 0; k0 < K; k0 += BK) {
      if (k0 + BK < K) {
        ISSUE_ASYNC(buf ? off0 : off1, Xrow + k0 + BK);
        asm volatile("s_wait_asynccnt 1" ::: "memory");
      } else {
        asm volatile("s_wait_asynccnt 0" ::: "memory");
      }
      __syncthreads();

      const __bf16 (*Ab)[APAD] = As[buf];
      v16bf a0 = *(const v16bf*)&Ab[wm * 32 + row16][khalf * 16];
      v16bf a1 = *(const v16bf*)&Ab[wm * 32 + 16 + row16][khalf * 16];
      v16bf b  = *(const v16bf*)&Wrow[k0 + khalf * 16];
      __builtin_prefetch((const void*)&Wrow[k0 + khalf * 16 + BK], 0, 3);

      acc0 = __builtin_amdgcn_wmma_f32_16x16x32_bf16(false, a0, false, b,
                                                     (short)0, acc0, false, false);
      acc1 = __builtin_amdgcn_wmma_f32_16x16x32_bf16(false, a1, false, b,
                                                     (short)0, acc1, false, false);
      __syncthreads();
      buf ^= 1;
    }
  } else {
    for (int k0 = 0; k0 < K; k0 += BK) {
      if (k0 + BK < K) {
        ISSUE_ASYNC(buf ? off0 : off1, Xrow + k0 + BK);
        asm volatile("s_wait_asynccnt 1" ::: "memory");
      } else {
        asm volatile("s_wait_asynccnt 0" ::: "memory");
      }
      __syncthreads();

      const __bf16 (*Ab)[APAD] = As[buf];
      v16bf a0 = *(const v16bf*)&Ab[wm * 32 + row16][khalf * 16];
      v16bf a1 = *(const v16bf*)&Ab[wm * 32 + 16 + row16][khalf * 16];

      v16bf b;
      int kb = k0 + khalf * 16;
      if (col < N && (kb + 16) <= K) {
        b = *(const v16bf*)&Wrow[kb];
      } else {
        union { v16bf v; unsigned u[8]; } z;
#pragma unroll
        for (int j = 0; j < 8; ++j) z.u[j] = 0u;
        b = z.v;
      }

      acc0 = __builtin_amdgcn_wmma_f32_16x16x32_bf16(false, a0, false, b,
                                                     (short)0, acc0, false, false);
      acc1 = __builtin_amdgcn_wmma_f32_16x16x32_bf16(false, a1, false, b,
                                                     (short)0, acc1, false, false);
      __syncthreads();
      buf ^= 1;
    }
  }
#undef ISSUE_ASYNC

  // --- store (lanes 0-15 -> rows v, lanes 16-31 -> rows v+8) ---
  if (col < N) {
    float bv = bias ? bias[col] : 0.0f;
    long r0 = blockM + wm * 32 + khalf * 8;
#pragma unroll
    for (int v = 0; v < 8; ++v) {
      float v0 = acc0[v] + bv;
      float v1 = acc1[v] + bv;
      long ra = r0 + v, rb = r0 + 16 + v;
      if (C)  { C[(size_t)ra * ldc + col] = v0;  C[(size_t)rb * ldc + col] = v1; }
      if (Cb) { Cb[(size_t)ra * ldcb + col] = f2bf(v0);
                Cb[(size_t)rb * ldcb + col] = f2bf(v1); }
    }
  }
}

// ---------------- feature encoder stage 1: lin(3->128)+LN+GELU -> bf16 -----
__global__ __launch_bounds__(256)
void fe_mlp1_k(const float* __restrict__ feat, const float* __restrict__ w1,
               const float* __restrict__ b1, const float* __restrict__ g1,
               const float* __restrict__ be1, __bf16* __restrict__ h1b) {
  int wid = threadIdx.x >> 5, lane = threadIdx.x & 31;
  long r = (long)blockIdx.x * 8 + wid;
  if (r >= (long)GN_ROWS) return;
  float x0 = feat[r * 3 + 0], x1 = feat[r * 3 + 1], x2 = feat[r * 3 + 2];
  float v[4], s = 0.f, sq = 0.f;
#pragma unroll
  for (int j = 0; j < 4; ++j) {
    int c = lane + 32 * j;
    float t = b1[c] + w1[c * 3] * x0 + w1[c * 3 + 1] * x1 + w1[c * 3 + 2] * x2;
    v[j] = t; s += t; sq += t * t;
  }
#pragma unroll
  for (int o = 16; o > 0; o >>= 1) { s += __shfl_xor(s, o, 32); sq += __shfl_xor(sq, o, 32); }
  float m = s * (1.0f / 128.0f);
  float inv = rsqrtf(sq * (1.0f / 128.0f) - m * m + 1e-5f);
#pragma unroll
  for (int j = 0; j < 4; ++j) {
    int c = lane + 32 * j;
    float y = (v[j] - m) * inv * g1[c] + be1[c];
    h1b[(size_t)r * 128 + c] = f2bf(gelu_f(y));
  }
}

// ---------------- pos encoder stage 1: lin(3->256)+GELU -> bf16 ------------
__global__ __launch_bounds__(256)
void pe_mlp1_k(const float* __restrict__ coord, const float* __restrict__ w1,
               const float* __restrict__ b1, __bf16* __restrict__ p1b) {
  int wid = threadIdx.x >> 5, lane = threadIdx.x & 31;
  long r = (long)blockIdx.x * 8 + wid;
  if (r >= (long)G_GROUPS) return;
  float x0 = coord[r * 3 + 0], x1 = coord[r * 3 + 1], x2 = coord[r * 3 + 2];
#pragma unroll
  for (int j = 0; j < 8; ++j) {
    int c = lane + 32 * j;
    float t = b1[c] + w1[c * 3] * x0 + w1[c * 3 + 1] * x1 + w1[c * 3 + 2] * x2;
    p1b[(size_t)r * 256 + c] = f2bf(gelu_f(t));
  }
}

// ---------------- generic LayerNorm (optional residual/GELU, f32+bf16 out) -
__global__ __launch_bounds__(256)
void layernorm_k(const float* __restrict__ x, int ldx,
                 const float* __restrict__ res, int ldr,
                 float* __restrict__ y, int ldy,
                 __bf16* __restrict__ yb, int ldyb,
                 const float* __restrict__ g, const float* __restrict__ b,
                 long M, int D, int act) {
  int wid = threadIdx.x >> 5, lane = threadIdx.x & 31;
  long r = (long)blockIdx.x * 8 + wid;
  if (r >= M) return;
  float v[16], s = 0.f, sq = 0.f;
#pragma unroll
  for (int i = 0; i < 16; ++i) {
    int c = lane + 32 * i;
    if (c < D) {
      float t = x[(size_t)r * ldx + c];
      if (res) t += res[(size_t)r * ldr + c];
      v[i] = t; s += t; sq += t * t;
    }
  }
#pragma unroll
  for (int o = 16; o > 0; o >>= 1) { s += __shfl_xor(s, o, 32); sq += __shfl_xor(sq, o, 32); }
  float invD = 1.0f / (float)D;
  float m = s * invD;
  float inv = rsqrtf(sq * invD - m * m + 1e-5f);
#pragma unroll
  for (int i = 0; i < 16; ++i) {
    int c = lane + 32 * i;
    if (c < D) {
      float o = (v[i] - m) * inv * g[c] + b[c];
      if (act) o = gelu_f(o);
      if (y)  y[(size_t)r * ldy + c] = o;
      if (yb) yb[(size_t)r * ldyb + c] = f2bf(o);
    }
  }
}

// ---------------- neighbor max + broadcast into concat half (bf16) ---------
__global__ void nbrmax_k(__bf16* __restrict__ hcatb) {
  long id = (long)blockIdx.x * blockDim.x + threadIdx.x;
  if (id >= (long)G_GROUPS * 256) return;
  long g = id >> 8; int c = (int)(id & 255);
  __bf16* base = hcatb + (size_t)g * 8 * 512;
  float m = bf2f(base[c]);
#pragma unroll
  for (int n = 1; n < 8; ++n) m = fmaxf(m, bf2f(base[(size_t)n * 512 + c]));
  __bf16 mb = f2bf(m);
#pragma unroll
  for (int n = 0; n < 8; ++n) base[(size_t)n * 512 + 256 + c] = mb;
}

// ---------------- max over 8 neighbors, f32 in -> strided bf16 out ---------
__global__ void rowmax8_k(const float* __restrict__ x, int ldx,
                          __bf16* __restrict__ y, int ldy) {
  long id = (long)blockIdx.x * blockDim.x + threadIdx.x;
  if (id >= (long)G_GROUPS * 256) return;
  long g = id >> 8; int c = (int)(id & 255);
  const float* p = x + (size_t)g * 8 * ldx + c;
  float m = p[0];
#pragma unroll
  for (int n = 1; n < 8; ++n) m = fmaxf(m, p[(size_t)n * ldx]);
  y[(size_t)g * ldy + c] = f2bf(m);
}

// ---------------- 2-level row gather (f32 in; f32 and/or bf16 out) ---------
__global__ void gather_rows_k(float* __restrict__ dst, int ldd,
                              __bf16* __restrict__ dstb, int lddb,
                              const float* __restrict__ src, int ldsv,
                              const int* __restrict__ idxA,
                              const int* __restrict__ idxB,
                              long M, int D) {
  long id = (long)blockIdx.x * blockDim.x + threadIdx.x;
  if (id >= M * D) return;
  long r = id / D; int c = (int)(id % D);
  int j = idxA ? idxA[r] : (int)r;
  if (idxB) j = idxB[j];
  float v = src[(size_t)j * ldsv + c];
  if (dst)  dst[(size_t)r * ldd + c] = v;
  if (dstb) dstb[(size_t)r * lddb + c] = f2bf(v);
}

// ---------------- causal depthwise conv1d (K=4) + SiLU ---------------------
__global__ void conv_silu_k(const float* __restrict__ xz,
                            const float* __restrict__ cw,
                            const float* __restrict__ cb,
                            float* __restrict__ xc, __bf16* __restrict__ xcb) {
  long id = (long)blockIdx.x * blockDim.x + threadIdx.x;
  if (id >= (long)G_GROUPS * D_INNER) return;
  long g = id >> 9; int d = (int)(id & 511);
  int t = (int)(g & (PATCH_L - 1));
  long base = g - t;                       // patch start row
  float s = cb[d];
#pragma unroll
  for (int k = 0; k < D_CONV; ++k) {
    int tt = t - (D_CONV - 1) + k;
    if (tt >= 0) s += cw[d * D_CONV + k] * xz[(size_t)(base + tt) * 1024 + d];
  }
  float o = silu_f(s);
  xc[id] = o;
  xcb[id] = f2bf(o);
}

// ---------------- softplus in place ----------------------------------------
__global__ void softplus_k(float* __restrict__ x, long n) {
  long id = (long)blockIdx.x * blockDim.x + threadIdx.x;
  if (id >= n) return;
  float v = x[id];
  x[id] = (v > 20.0f) ? v : log1pf(__expf(v));
}

// ---------------- selective scan: one patch per block, one channel per lane -
__global__ __launch_bounds__(512)
void scan_k(const float* __restrict__ dt, const float* __restrict__ xc,
            const float* __restrict__ xdbl, const float* __restrict__ A_log,
            float* __restrict__ ys) {
  __shared__ float Bs[D_STATE], Cs[D_STATE];
  int p = blockIdx.x, d = threadIdx.x;
  float Aa[D_STATE], h[D_STATE];
#pragma unroll
  for (int s = 0; s < D_STATE; ++s) {
    Aa[s] = -__expf(A_log[d * D_STATE + s]);
    h[s] = 0.0f;
  }
  for (int t = 0; t < PATCH_L; ++t) {
    long row = (long)p * PATCH_L + t;
    if (d < D_STATE)            Bs[d] = xdbl[(size_t)row * 48 + 16 + d];
    else if (d < 2 * D_STATE)   Cs[d - D_STATE] = xdbl[(size_t)row * 48 + 32 + (d - D_STATE)];
    __syncthreads();
    float dtv = dt[(size_t)row * D_INNER + d];
    float xv  = xc[(size_t)row * D_INNER + d];
    float dx  = dtv * xv;
    float y = 0.0f;
#pragma unroll
    for (int s = 0; s < D_STATE; ++s) {
      h[s] = h[s] * __expf(dtv * Aa[s]) + dx * Bs[s];
      y += h[s] * Cs[s];
    }
    ys[(size_t)row * D_INNER + d] = y;
    __syncthreads();
  }
}

// ---------------- gate: ys = (ys + D*xc) * silu(z); also bf16 copy ----------
__global__ void gate_k(float* __restrict__ ys, __bf16* __restrict__ ysb,
                       const float* __restrict__ xc,
                       const float* __restrict__ xz, const float* __restrict__ Dp) {
  long id = (long)blockIdx.x * blockDim.x + threadIdx.x;
  if (id >= (long)G_GROUPS * D_INNER) return;
  long g = id >> 9; int d = (int)(id & 511);
  float z = xz[(size_t)g * 1024 + 512 + d];
  float y = (ys[id] + Dp[d] * xc[id]) * silu_f(z);
  ys[id] = y;
  ysb[id] = f2bf(y);
}

// ============================ host side ====================================

static inline int eb(long n) { return (int)((n + 255) / 256); }

static void gemm(hipStream_t s, const __bf16* X, int lda, const __bf16* W,
                 const float* bias, float* C, int ldc, __bf16* Cb, int ldcb,
                 long M, int N, int K) {
  dim3 grid((N + BN - 1) / BN, (unsigned)((M + BM - 1) / BM));
  wmma_gemm_bf16_k<<<grid, 256, 0, s>>>(X, lda, W, bias, C, ldc, Cb, ldcb, M, N, K);
}
static void ln(hipStream_t s, const float* x, int ldx, const float* res, int ldr,
               float* y, int ldy, __bf16* yb, int ldyb,
               const float* g, const float* b, long M, int D, int act) {
  layernorm_k<<<dim3((unsigned)((M + 7) / 8)), 256, 0, s>>>(
      x, ldx, res, ldr, y, ldy, yb, ldyb, g, b, M, D, act);
}

extern "C" void kernel_launch(void* const* d_in, const int* in_sizes, int n_in,
                              void* d_out, int out_size, void* d_ws, size_t ws_size,
                              hipStream_t stream) {
  (void)in_sizes; (void)n_in; (void)out_size; (void)ws_size;
  // ---- inputs (setup_inputs order; params tree-flattened, keys sorted) ----
#define FI(i) ((const float*)d_in[(i)])
#define II(i) ((const int*)d_in[(i)])
  const float* feat       = FI(0);
  const float* coord      = FI(1);
  const int* order_idx    = II(2);
  const int* inverse_idx  = II(3);
  const int* pad_idx      = II(4);
  const int* unpad_idx    = II(5);
  const int* shift        = II(6);
  const int* shift_back   = II(7);
  // fe
  const float* fe_b1 = FI(8),  *fe_b2 = FI(9),  *fe_b3 = FI(10), *fe_b4 = FI(11);
  const float* ln1_b = FI(12), *ln1_g = FI(13), *ln2_b = FI(14), *ln2_g = FI(15);
  const float* fe_w1 = FI(16), *fe_w2 = FI(17), *fe_w3 = FI(18), *fe_w4 = FI(19);
  // pe
  const float* pe_b1 = FI(20), *pe_b2 = FI(21), *pe_w1 = FI(22), *pe_w2 = FI(23);
  // swin0.m0 / m1 (A_log, D, conv_b, conv_w, dt_proj_b, dt_proj_w, in_proj_w, out_proj_w, x_proj_w)
  const float* A_log[2]  = {FI(24), FI(33)};
  const float* Dp[2]     = {FI(25), FI(34)};
  const float* conv_b[2] = {FI(26), FI(35)};
  const float* conv_w[2] = {FI(27), FI(36)};
  const float* dt_b[2]   = {FI(28), FI(37)};
  const float* dt_w[2]   = {FI(29), FI(38)};
  const float* in_w[2]   = {FI(30), FI(39)};
  const float* out_w[2]  = {FI(31), FI(40)};
  const float* xp_w[2]   = {FI(32), FI(41)};
  const float* merge_b = FI(42), *merge_w = FI(43);
  const float* nfb[2]   = {FI(44), FI(46)};
  const float* nfg[2]   = {FI(45), FI(47)};
  const float* nf_b = FI(48), *nf_g = FI(49);
  const float* normb[2] = {FI(50), FI(52)};
  const float* normg[2] = {FI(51), FI(53)};
  const float* tok_b = FI(54), *tok_w = FI(55);

  // ---- workspace bump allocator (256B aligned) ----
  char* base = (char*)d_ws;
  size_t off = 0;
  auto takeB = [&](size_t bytes) -> void* {
    void* p = base + off;
    off += (bytes + 255) & ~(size_t)255;
    return p;
  };
  auto takeF = [&](long n) -> float* { return (float*)takeB((size_t)n * 4); };
  auto takeH = [&](long n) -> __bf16* { return (__bf16*)takeB((size_t)n * 2); };

  // bf16 weight pool
  __bf16* w2b   = takeH(256L * 128);
  __bf16* w3b   = takeH(512L * 512);
  __bf16* w4b   = takeH(256L * 512);
  __bf16* pw2b  = takeH(256L * 256);
  __bf16* tokb  = takeH(256L * 512);
  __bf16* inpb[2]  = {takeH(1024L * 256), takeH(1024L * 256)};
  __bf16* xpb[2]   = {takeH(48L * 512),   takeH(48L * 512)};
  __bf16* dtpb[2]  = {takeH(512L * 16),   takeH(512L * 16)};
  __bf16* outpb[2] = {takeH(256L * 512),  takeH(256L * 512)};
  __bf16* mrgb  = takeH(256L * 512);

  // activations
  __bf16* h1b   = takeH((long)GN_ROWS * 128);
  __bf16* hcatb = takeH((long)GN_ROWS * 512);
  float*  t3    = takeF((long)GN_ROWS * 512);
  __bf16* t3b   = takeH((long)GN_ROWS * 512);
  float*  t4    = takeF((long)GN_ROWS * 256);
  __bf16* cat0b = takeH((long)G_GROUPS * 512);
  __bf16* p1b   = takeH((long)G_GROUPS * 256);
  float*  f0    = takeF((long)G_GROUPS * 256);
  float*  fbuf  = takeF((long)G_GROUPS * 256);
  float*  xin1  = takeF((long)G_GROUPS * 256);
  __bf16* xnb   = takeH((long)G_GROUPS * 256);
  float*  xz    = takeF((long)G_GROUPS * 1024);
  float*  xc    = takeF((long)G_GROUPS * 512);
  __bf16* xcb   = takeH((long)G_GROUPS * 512);
  float*  xdbl  = takeF((long)G_GROUPS * 48);
  __bf16* xdblb = takeH((long)G_GROUPS * 48);
  float*  dtb   = takeF((long)G_GROUPS * 512);
  float*  ysb   = takeF((long)G_GROUPS * 512);
  __bf16* ysbf  = takeH((long)G_GROUPS * 512);
  float*  yob   = takeF((long)G_GROUPS * 256);
  float*  hbr   = takeF((long)G_GROUPS * 256);
  __bf16* cat1b = takeH((long)G_GROUPS * 512);
  float*  fm    = takeF((long)G_GROUPS * 256);

  // ---- pack weights to bf16 ----
  struct PackJob { const float* s; __bf16* d; long n; };
  PackJob jobs[] = {
    {fe_w2, w2b, 256L * 128}, {fe_w3, w3b, 512L * 512}, {fe_w4, w4b, 256L * 512},
    {pe_w2, pw2b, 256L * 256}, {tok_w, tokb, 256L * 512},
    {in_w[0], inpb[0], 1024L * 256}, {in_w[1], inpb[1], 1024L * 256},
    {xp_w[0], xpb[0], 48L * 512},    {xp_w[1], xpb[1], 48L * 512},
    {dt_w[0], dtpb[0], 512L * 16},   {dt_w[1], dtpb[1], 512L * 16},
    {out_w[0], outpb[0], 256L * 512},{out_w[1], outpb[1], 256L * 512},
    {merge_w, mrgb, 256L * 512},
  };
  for (auto& j : jobs)
    pack_bf16_k<<<eb(j.n), 256, 0, stream>>>(j.s, j.d, j.n);

  // ---- feature encoder ----
  fe_mlp1_k<<<GN_ROWS / 8, 256, 0, stream>>>(feat, fe_w1, fe_b1, ln1_g, ln1_b, h1b);
  gemm(stream, h1b, 128, w2b, fe_b2, nullptr, 0, hcatb, 512, GN_ROWS, 256, 128);
  nbrmax_k<<<eb((long)G_GROUPS * 256), 256, 0, stream>>>(hcatb);
  gemm(stream, hcatb, 512, w3b, fe_b3, t3, 512, nullptr, 0, GN_ROWS, 512, 512);
  ln(stream, t3, 512, nullptr, 0, nullptr, 0, t3b, 512, ln2_g, ln2_b, GN_ROWS, 512, 1);
  gemm(stream, t3b, 512, w4b, fe_b4, t4, 256, nullptr, 0, GN_ROWS, 256, 512);
  rowmax8_k<<<eb((long)G_GROUPS * 256), 256, 0, stream>>>(t4, 256, cat0b, 512);

  // ---- pos encoder -> second half of cat0b ----
  pe_mlp1_k<<<G_GROUPS / 8, 256, 0, stream>>>(coord, pe_w1, pe_b1, p1b);
  gemm(stream, p1b, 256, pw2b, pe_b2, nullptr, 0, cat0b + 256, 512, G_GROUPS, 256, 256);

  // ---- token projection + serialize-order gather (f = f0[order[pad]]) ----
  gemm(stream, cat0b, 512, tokb, tok_b, f0, 256, nullptr, 0, G_GROUPS, 256, 512);
  gather_rows_k<<<eb((long)G_GROUPS * 256), 256, 0, stream>>>(
      fbuf, 256, nullptr, 0, f0, 256, pad_idx, order_idx, G_GROUPS, 256);

  // ---- two mamba branches ----
  for (int br = 0; br < 2; ++br) {
    const float* xin = fbuf;
    if (br == 1) {
      gather_rows_k<<<eb((long)G_GROUPS * 256), 256, 0, stream>>>(
          xin1, 256, nullptr, 0, fbuf, 256, shift, nullptr, G_GROUPS, 256);
      xin = xin1;
    }
    ln(stream, xin, 256, nullptr, 0, nullptr, 0, xnb, 256, normg[br], normb[br],
       G_GROUPS, 256, 0);
    gemm(stream, xnb, 256, inpb[br], nullptr, xz, 1024, nullptr, 0, G_GROUPS, 1024, 256);
    conv_silu_k<<<eb((long)G_GROUPS * 512), 256, 0, stream>>>(xz, conv_w[br], conv_b[br], xc, xcb);
    gemm(stream, xcb, 512, xpb[br], nullptr, xdbl, 48, xdblb, 48, G_GROUPS, 48, 512);
    gemm(stream, xdblb, 48, dtpb[br], dt_b[br], dtb, 512, nullptr, 0, G_GROUPS, 512, 16);
    softplus_k<<<eb((long)G_GROUPS * 512), 256, 0, stream>>>(dtb, (long)G_GROUPS * 512);
    scan_k<<<N_PATCH, 512, 0, stream>>>(dtb, xc, xdbl, A_log[br], ysb);
    gate_k<<<eb((long)G_GROUPS * 512), 256, 0, stream>>>(ysb, ysbf, xc, xz, Dp[br]);
    gemm(stream, ysbf, 512, outpb[br], nullptr, yob, 256, nullptr, 0, G_GROUPS, 256, 512);
    if (br == 0) {
      ln(stream, yob, 256, xin, 256, nullptr, 0, cat1b, 512, nfg[0], nfb[0],
         G_GROUPS, 256, 0);
    } else {
      ln(stream, yob, 256, xin, 256, hbr, 256, nullptr, 0, nfg[1], nfb[1],
         G_GROUPS, 256, 0);
      gather_rows_k<<<eb((long)G_GROUPS * 256), 256, 0, stream>>>(
          nullptr, 0, cat1b + 256, 512, hbr, 256, shift_back, nullptr, G_GROUPS, 256);
    }
  }

  // ---- merge + final LN + inverse scatter ----
  gemm(stream, cat1b, 512, mrgb, merge_b, fm, 256, nullptr, 0, G_GROUPS, 256, 512);
  ln(stream, fm, 256, nullptr, 0, fm, 256, nullptr, 0, nf_g, nf_b, G_GROUPS, 256, 0);
  gather_rows_k<<<eb((long)G_GROUPS * 256), 256, 0, stream>>>(
      (float*)d_out, 256, nullptr, 0, fm, 256, inverse_idx, unpad_idx, G_GROUPS, 256);
}
